// QuantizedLinear_15006615734457
// MI455X (gfx1250) — compile-verified, matrix operand order checked by reading, not run on hardware
//
#include <hip/hip_runtime.h>
#include <stdint.h>

// CDNA5 / gfx1250 quantized-linear GEMM:
//   out[m, n] = scale[n] * ( sum_k x[m,k] * wq[n,k] ) + bias[n]
//
// Pass 1: one-shot convert x (f32) and wq (i32, exact in f16) to f16 in d_ws.
// Pass 2: f16 WMMA GEMM; tiles staged into LDS by the Tensor Data Mover
//         (TENSOR_LOAD_TO_LDS descriptors, TENSORcnt double buffering) when
//         the builtin exists, else per-lane async global->LDS (ASYNCcnt).
// Fallback (ws too small): fused-convert single-pass kernel.

typedef _Float16 v16h __attribute__((ext_vector_type(16)));
typedef _Float16 v8h  __attribute__((ext_vector_type(8)));
typedef float    v8f  __attribute__((ext_vector_type(8)));
typedef unsigned int u32x4 __attribute__((ext_vector_type(4)));
typedef int          i32x8 __attribute__((ext_vector_type(8)));
typedef int          i32x4 __attribute__((ext_vector_type(4)));

#define BM 128
#define BN 128
#define THREADS 256

#if defined(__has_builtin)
#if __has_builtin(__builtin_amdgcn_tensor_load_to_lds)
#define USE_TDM 1
#endif
#endif

__device__ __forceinline__ void async_ld_b128(uint32_t lds_off, const void* g) {
  // LDS[lds_off .. +15] = MEM[g .. +15], tracked by ASYNCcnt
  asm volatile("global_load_async_to_lds_b128 %0, %1, off"
               :: "v"(lds_off), "v"((uint64_t)g)
               : "memory");
}

#ifdef USE_TDM
// Issue one TDM 2D tile load: tile_d1 rows x tile_d0 f16, row stride = stride_d0
// f16 in global memory, destination LDS rows padded by 4 dwords every 32 dwords
// (pad_interval=4 -> 1<<4 8B units = 128B, pad_amount=3 -> 4 dwords).
// D# layout per CDNA5 ISA (group0 128b, group1 256b).
// This toolchain's builtin is the 6-arg clang-23 form:
//   (u32x4 g0, i32x8 g1, i32x4 g2, i32x4 g3, i32x8 extra, i32 cpol)
__device__ __forceinline__ void tdm_load_2d_f16(uint32_t lds_byte, const void* gaddr,
                                                int tensor_d0, int tensor_d1,
                                                int tile_d0, int tile_d1,
                                                int stride_d0) {
  const uint64_t ga = (uint64_t)gaddr;
  u32x4 g0;
  g0[0] = 1u;                                              // count=1 (valid), user mode
  g0[1] = lds_byte;                                        // lds_addr
  g0[2] = (uint32_t)ga;                                    // global_addr[31:0]
  g0[3] = (uint32_t)((ga >> 32) & 0x1FFFFFFu) | (2u << 30);// global_addr[56:32], type=2
  i32x8 g1;
  g1[0] = (1 << 16)        // data_size = 2 bytes
        | (1 << 20)        // pad_enable
        | (4 << 22)        // pad_interval: 1<<4 8B units = 32 dwords
        | (3 << 25);       // pad_amount: 4 dwords
  g1[1] = (tensor_d0 & 0xFFFF) << 16;                            // tensor_dim0[15:0]
  g1[2] = ((tensor_d0 >> 16) & 0xFFFF) | ((tensor_d1 & 0xFFFF) << 16); // dim0 hi | dim1 lo
  g1[3] = ((tensor_d1 >> 16) & 0xFFFF) | ((tile_d0 & 0xFFFF) << 16);   // dim1 hi | tile_dim0
  g1[4] = tile_d1 & 0xFFFF;                                      // tile_dim1 (tile_dim2=0)
  g1[5] = stride_d0;                                             // tensor_dim0_stride[31:0]
  g1[6] = 0;                                                     // stride hi | dim1_stride lo
  g1[7] = 0;
  const i32x4 z4 = {0, 0, 0, 0};                                 // groups 2/3 unused (2D)
  const i32x8 z8 = {0, 0, 0, 0, 0, 0, 0, 0};
  __builtin_amdgcn_tensor_load_to_lds(g0, g1, z4, z4, z8, 0);
}
#endif

// ---------------------------------------------------------------------------
// Pre-pass converters (each element touched exactly once on the whole device)
// ---------------------------------------------------------------------------
__global__ __launch_bounds__(THREADS)
void cvt_f32_to_f16_kernel(const float* __restrict__ in, _Float16* __restrict__ out, size_t n) {
  size_t i = ((size_t)blockIdx.x * THREADS + threadIdx.x) * 8;
  if (i >= n) return;
  const float4 a = *(const float4*)(in + i);
  const float4 b = *(const float4*)(in + i + 4);
  v8h v;
  v[0] = (_Float16)a.x; v[1] = (_Float16)a.y; v[2] = (_Float16)a.z; v[3] = (_Float16)a.w;
  v[4] = (_Float16)b.x; v[5] = (_Float16)b.y; v[6] = (_Float16)b.z; v[7] = (_Float16)b.w;
  *(v8h*)(out + i) = v;
}

__global__ __launch_bounds__(THREADS)
void cvt_i32_to_f16_kernel(const int* __restrict__ in, _Float16* __restrict__ out, size_t n) {
  size_t i = ((size_t)blockIdx.x * THREADS + threadIdx.x) * 8;
  if (i >= n) return;
  const int4 a = *(const int4*)(in + i);
  const int4 b = *(const int4*)(in + i + 4);
  v8h v;
  v[0] = (_Float16)a.x; v[1] = (_Float16)a.y; v[2] = (_Float16)a.z; v[3] = (_Float16)a.w;
  v[4] = (_Float16)b.x; v[5] = (_Float16)b.y; v[6] = (_Float16)b.z; v[7] = (_Float16)b.w;
  *(v8h*)(out + i) = v;
}

// ---------------------------------------------------------------------------
// Main path: pure-f16 WMMA GEMM
// ---------------------------------------------------------------------------
#define BK_H 64        // f16 K-elements per LDS stage = 2 WMMA K-steps
#define STRIDE_H 72    // 64 + 8 pad f16 -> row stride 144B = 36 banks, conflict-free

// A-fragment slice per ISA 7.12.2 (16-bit A 16x32): lane half h holds
// elems 0..7 -> K=h*8+j, elems 8..15 -> K=16+h*8+j (plus ks*32 step offset).
__device__ __forceinline__ v16h load_frag(const _Float16* row, int ks, int half) {
  const v8h lo = *(const v8h*)(row + ks * 32 + half * 8);
  const v8h hi = *(const v8h*)(row + ks * 32 + 16 + half * 8);
  return __builtin_shufflevector(lo, hi, 0, 1, 2, 3, 4, 5, 6, 7,
                                         8, 9, 10, 11, 12, 13, 14, 15);
}

__global__ __launch_bounds__(THREADS)
void qlinear_wmma_f16_kernel(const _Float16* __restrict__ xh,
                             const _Float16* __restrict__ wh,
                             const float* __restrict__ scale,
                             const float* __restrict__ bias,
                             float*       __restrict__ out,
                             int M, int N, int K)
{
  __shared__ _Float16 Ah[2][BM][STRIDE_H];
  __shared__ _Float16 Wh[2][BN][STRIDE_H];

  const int tid  = threadIdx.x;
  const int lane = tid & 31;
  const int wave = tid >> 5;     // 8 waves
  const int wm   = wave & 1;     // 2 wave rows  -> 64 M rows each
  const int wn   = wave >> 1;    // 4 wave cols  -> 32 N cols each
  const int l16  = lane & 15;
  const int half = lane >> 4;

  const int n_blk = blockIdx.x * BN;
  const int m_blk = blockIdx.y * BM;

  const uint32_t a_lds0 = (uint32_t)(uintptr_t)&Ah[0][0][0];
  const uint32_t w_lds0 = (uint32_t)(uintptr_t)&Wh[0][0][0];

  const int KT = K / BK_H;

#ifdef USE_TDM
  // One TDM descriptor per matrix per stage: 128 rows x 64 f16 tile, hardware
  // pads LDS rows to 144B. Issued by wave 0 only (EXEC ignored by TDM).
  auto issue_stage = [&](int kb, int buf) {
    const size_t k0 = (size_t)kb * BK_H;
    tdm_load_2d_f16(a_lds0 + (uint32_t)(buf * BM * STRIDE_H * 2),
                    (const void*)(xh + (size_t)m_blk * K + k0),
                    K, M, BK_H, BM, K);
    tdm_load_2d_f16(w_lds0 + (uint32_t)(buf * BN * STRIDE_H * 2),
                    (const void*)(wh + (size_t)n_blk * K + k0),
                    K, N, BK_H, BN, K);
  };
#else
  // Stage = 128 rows x 128B (64 f16) per matrix: 1024 16B segments each,
  // 4 A + 4 W async b128 loads per thread, consecutive tids -> consecutive 16B.
  auto issue_stage = [&](int kb, int buf) {
    const size_t k0 = (size_t)kb * BK_H;
#pragma unroll
    for (int i = 0; i < 4; ++i) {
      const int seg = i * THREADS + tid;   // 0..1023
      const int row = seg >> 3;            // 0..127
      const int c8  = (seg & 7) * 8;       // f16 col (16B steps)
      async_ld_b128(a_lds0 + (uint32_t)(((buf * BM + row) * STRIDE_H + c8) * 2),
                    (const void*)(xh + (size_t)(m_blk + row) * K + k0 + c8));
    }
#pragma unroll
    for (int i = 0; i < 4; ++i) {
      const int seg = i * THREADS + tid;
      const int row = seg >> 3;
      const int c8  = (seg & 7) * 8;
      async_ld_b128(w_lds0 + (uint32_t)(((buf * BN + row) * STRIDE_H + c8) * 2),
                    (const void*)(wh + (size_t)(n_blk + row) * K + k0 + c8));
    }
  };
#endif

  v8f acc[4][2];
#pragma unroll
  for (int mt = 0; mt < 4; ++mt)
#pragma unroll
    for (int nt = 0; nt < 2; ++nt)
#pragma unroll
      for (int r = 0; r < 8; ++r) acc[mt][nt][r] = 0.0f;

#ifdef USE_TDM
  if (wave == 0) issue_stage(0, 0);
#else
  issue_stage(0, 0);
#endif
  int buf = 0;

  for (int kb = 0; kb < KT; ++kb) {
#ifdef USE_TDM
    if (kb + 1 < KT) {
      if (wave == 0) issue_stage(kb + 1, buf ^ 1);       // prefetch next stage
      __builtin_amdgcn_s_wait_tensorcnt(2);              // current stage landed
    } else {
      __builtin_amdgcn_s_wait_tensorcnt(0);
    }
#else
    if (kb + 1 < KT) {
      issue_stage(kb + 1, buf ^ 1);                      // prefetch next stage
      asm volatile("s_wait_asynccnt 8" ::: "memory");    // current stage landed
    } else {
      asm volatile("s_wait_asynccnt 0" ::: "memory");
    }
#endif
    __syncthreads();   // DMA for `buf` visible to all waves

#pragma unroll
    for (int ks = 0; ks < 2; ++ks) {                     // 2 WMMA K-steps / stage
      v16h afrag[4], bfrag[2];
#pragma unroll
      for (int mt = 0; mt < 4; ++mt)
        afrag[mt] = load_frag(&Ah[buf][wm * 64 + mt * 16 + l16][0], ks, half);
#pragma unroll
      for (int nt = 0; nt < 2; ++nt)
        bfrag[nt] = load_frag(&Wh[buf][wn * 32 + nt * 16 + l16][0], ks, half);

#pragma unroll
      for (int mt = 0; mt < 4; ++mt)
#pragma unroll
        for (int nt = 0; nt < 2; ++nt)
          acc[mt][nt] = __builtin_amdgcn_wmma_f32_16x16x32_f16(
              false, afrag[mt], false, bfrag[nt],
              (short)0, acc[mt][nt], false, false);
    }

    __syncthreads();   // done reading `buf`
    buf ^= 1;
  }

  // Epilogue: D layout: lane 0-15 -> M=r, N=l16 ; lane 16-31 -> M=8+r.
#pragma unroll
  for (int nt = 0; nt < 2; ++nt) {
    const int col = n_blk + wn * 32 + nt * 16 + l16;
    const float sc = scale[col];
    const float bs = bias[col];
#pragma unroll
    for (int mt = 0; mt < 4; ++mt) {
      const int row0 = m_blk + wm * 64 + mt * 16 + half * 8;
#pragma unroll
      for (int r = 0; r < 8; ++r)
        out[(size_t)(row0 + r) * N + col] = acc[mt][nt][r] * sc + bs;
    }
  }
}

// ---------------------------------------------------------------------------
// Fallback (ws too small): fused-convert single-pass kernel
// ---------------------------------------------------------------------------
#define BK_F 32
#define STRIDE_F 36

__device__ __forceinline__ v16h pack_a(const float* r, int half) {
  const float4 f0 = *(const float4*)(r + half * 8);
  const float4 f1 = *(const float4*)(r + half * 8 + 4);
  const float4 f2 = *(const float4*)(r + 16 + half * 8);
  const float4 f3 = *(const float4*)(r + 16 + half * 8 + 4);
  v16h v;
  v[0]  = (_Float16)f0.x; v[1]  = (_Float16)f0.y; v[2]  = (_Float16)f0.z; v[3]  = (_Float16)f0.w;
  v[4]  = (_Float16)f1.x; v[5]  = (_Float16)f1.y; v[6]  = (_Float16)f1.z; v[7]  = (_Float16)f1.w;
  v[8]  = (_Float16)f2.x; v[9]  = (_Float16)f2.y; v[10] = (_Float16)f2.z; v[11] = (_Float16)f2.w;
  v[12] = (_Float16)f3.x; v[13] = (_Float16)f3.y; v[14] = (_Float16)f3.z; v[15] = (_Float16)f3.w;
  return v;
}

__device__ __forceinline__ v16h pack_w(const int* r, int half) {
  const int4 i0 = *(const int4*)(r + half * 8);
  const int4 i1 = *(const int4*)(r + half * 8 + 4);
  const int4 i2 = *(const int4*)(r + 16 + half * 8);
  const int4 i3 = *(const int4*)(r + 16 + half * 8 + 4);
  v16h v;
  v[0]  = (_Float16)i0.x; v[1]  = (_Float16)i0.y; v[2]  = (_Float16)i0.z; v[3]  = (_Float16)i0.w;
  v[4]  = (_Float16)i1.x; v[5]  = (_Float16)i1.y; v[6]  = (_Float16)i1.z; v[7]  = (_Float16)i1.w;
  v[8]  = (_Float16)i2.x; v[9]  = (_Float16)i2.y; v[10] = (_Float16)i2.z; v[11] = (_Float16)i2.w;
  v[12] = (_Float16)i3.x; v[13] = (_Float16)i3.y; v[14] = (_Float16)i3.z; v[15] = (_Float16)i3.w;
  return v;
}

__global__ __launch_bounds__(THREADS)
void qlinear_wmma_fused_kernel(const float* __restrict__ x,
                               const int*   __restrict__ wq,
                               const float* __restrict__ scale,
                               const float* __restrict__ bias,
                               float*       __restrict__ out,
                               int M, int N, int K)
{
  __shared__ float Asm[2][BM][STRIDE_F];
  __shared__ int   Wsm[2][BN][STRIDE_F];

  const int tid  = threadIdx.x;
  const int lane = tid & 31;
  const int wave = tid >> 5;
  const int wm   = wave & 1;
  const int wn   = wave >> 1;
  const int l16  = lane & 15;
  const int half = lane >> 4;

  const int n_blk = blockIdx.x * BN;
  const int m_blk = blockIdx.y * BM;

  const uint32_t a_lds0 = (uint32_t)(uintptr_t)&Asm[0][0][0];
  const uint32_t w_lds0 = (uint32_t)(uintptr_t)&Wsm[0][0][0];

  const int KT = K / BK_F;

  auto issue_stage = [&](int kb, int buf) {
    const size_t k0 = (size_t)kb * BK_F;
#pragma unroll
    for (int i = 0; i < 4; ++i) {
      const int seg = i * THREADS + tid;
      const int row = seg >> 3;
      const int c4  = (seg & 7) * 4;
      async_ld_b128(a_lds0 + (uint32_t)(((buf * BM + row) * STRIDE_F + c4) * 4),
                    (const void*)(x + (size_t)(m_blk + row) * K + k0 + c4));
    }
#pragma unroll
    for (int i = 0; i < 4; ++i) {
      const int seg = i * THREADS + tid;
      const int row = seg >> 3;
      const int c4  = (seg & 7) * 4;
      async_ld_b128(w_lds0 + (uint32_t)(((buf * BN + row) * STRIDE_F + c4) * 4),
                    (const void*)(wq + (size_t)(n_blk + row) * K + k0 + c4));
    }
  };

  v8f acc[4][2];
#pragma unroll
  for (int mt = 0; mt < 4; ++mt)
#pragma unroll
    for (int nt = 0; nt < 2; ++nt)
#pragma unroll
      for (int r = 0; r < 8; ++r) acc[mt][nt][r] = 0.0f;

  issue_stage(0, 0);
  int buf = 0;

  for (int kb = 0; kb < KT; ++kb) {
    if (kb + 1 < KT) {
      issue_stage(kb + 1, buf ^ 1);
      asm volatile("s_wait_asynccnt 8" ::: "memory");
    } else {
      asm volatile("s_wait_asynccnt 0" ::: "memory");
    }
    __syncthreads();

    v16h afrag[4], bfrag[2];
#pragma unroll
    for (int mt = 0; mt < 4; ++mt)
      afrag[mt] = pack_a(&Asm[buf][wm * 64 + mt * 16 + l16][0], half);
#pragma unroll
    for (int nt = 0; nt < 2; ++nt)
      bfrag[nt] = pack_w(&Wsm[buf][wn * 32 + nt * 16 + l16][0], half);

#pragma unroll
    for (int mt = 0; mt < 4; ++mt)
#pragma unroll
      for (int nt = 0; nt < 2; ++nt)
        acc[mt][nt] = __builtin_amdgcn_wmma_f32_16x16x32_f16(
            false, afrag[mt], false, bfrag[nt],
            (short)0, acc[mt][nt], false, false);

    __syncthreads();
    buf ^= 1;
  }

#pragma unroll
  for (int nt = 0; nt < 2; ++nt) {
    const int col = n_blk + wn * 32 + nt * 16 + l16;
    const float sc = scale[col];
    const float bs = bias[col];
#pragma unroll
    for (int mt = 0; mt < 4; ++mt) {
      const int row0 = m_blk + wm * 64 + mt * 16 + half * 8;
#pragma unroll
      for (int r = 0; r < 8; ++r)
        out[(size_t)(row0 + r) * N + col] = acc[mt][nt][r] * sc + bs;
    }
  }
}

// ---------------------------------------------------------------------------
extern "C" void kernel_launch(void* const* d_in, const int* in_sizes, int n_in,
                              void* d_out, int out_size, void* d_ws, size_t ws_size,
                              hipStream_t stream) {
  const float* x     = (const float*)d_in[0];
  const int*   wq    = (const int*)d_in[1];
  const float* scale = (const float*)d_in[2];
  const float* bias  = (const float*)d_in[3];
  float*       out   = (float*)d_out;

  const int N = in_sizes[2];           // OUT = 16384
  const int K = in_sizes[1] / N;       // IN  = 4096
  const int M = in_sizes[0] / K;       // B*S = 16384

  const size_t x_elems = (size_t)in_sizes[0];
  const size_t w_elems = (size_t)in_sizes[1];
  const size_t need    = (x_elems + w_elems) * sizeof(_Float16);

  dim3 grid(N / BN, M / BM);           // (128, 128)

  if (ws_size >= need) {
    _Float16* xh = (_Float16*)d_ws;
    _Float16* wh = xh + x_elems;
    cvt_f32_to_f16_kernel<<<(unsigned)((x_elems + THREADS * 8 - 1) / (THREADS * 8)),
                            THREADS, 0, stream>>>(x, xh, x_elems);
    cvt_i32_to_f16_kernel<<<(unsigned)((w_elems + THREADS * 8 - 1) / (THREADS * 8)),
                            THREADS, 0, stream>>>(wq, wh, w_elems);
    qlinear_wmma_f16_kernel<<<grid, THREADS, 0, stream>>>(xh, wh, scale, bias, out, M, N, K);
  } else {
    qlinear_wmma_fused_kernel<<<grid, THREADS, 0, stream>>>(x, wq, scale, bias, out, M, N, K);
  }
}
